// MultiHeadAttention_79886391705915
// MI455X (gfx1250) — matrix operand order repaired and checked by reading, not compiled, not run on hardware
//
#include <hip/hip_runtime.h>
#include <hip/hip_bf16.h>

// ---------------------------------------------------------------------------
// MHA for MI455X (gfx1250): bf16 WMMA everywhere, fp32 only for softmax/attn.
// B=2, T=2048, D=1024, H=16, DH=64.
// d_out = [out (B*T*D fp32)] ++ [attn (B*H*T*T fp32)]
// Round 2: register-blocked GEMMs (32x64/wave), blocked attn@V (16x64/wave),
//          non-temporal streaming for the 537MB attn tensor.
// ---------------------------------------------------------------------------

#define B_ 2
#define T_ 2048
#define D_ 1024
#define H_ 16
#define DH_ 64
#define M_ (B_ * T_)          // 4096 rows for projection GEMMs

typedef __bf16 bf16_t;
typedef bf16_t bf16x16 __attribute__((ext_vector_type(16)));
typedef float  f32x8   __attribute__((ext_vector_type(8)));
typedef float  f32x4   __attribute__((ext_vector_type(4)));

union FragU { bf16x16 v; uint4 q[2]; };

#define WMMA_BF16(a, b, c) \
  __builtin_amdgcn_wmma_f32_16x16x32_bf16(false, (a), false, (b), (short)0, (c), false, false)

// A fragment: 16x32 (MxK), 16-bit. Lane L: row = L&15; k-chunks of 8 at
// klo and klo+16 where klo = (L<16)?0:8.  (ISA 7.12.2, 16-bit A 16x32)
__device__ __forceinline__ bf16x16 load_fragA(const bf16_t* tile, int ld, int lane) {
  const int r  = lane & 15;
  const int ko = (lane < 16) ? 0 : 8;
  const bf16_t* p = tile + (size_t)r * ld + ko;
  FragU f;
  f.q[0] = *(const uint4*)(p);
  f.q[1] = *(const uint4*)(p + 16);
  return f.v;
}

// B fragment loaded from B^T (row-major, K contiguous). Lane L: column
// n = L&15; k-chunks of 8 at klo and klo+16 where klo = (L<16)?0:8.
__device__ __forceinline__ bf16x16 load_fragB(const bf16_t* tileT, int ld, int lane) {
  const int c  = lane & 15;
  const int ko = (lane < 16) ? 0 : 8;
  const bf16_t* p = tileT + (size_t)c * ld + ko;
  FragU f;
  f.q[0] = *(const uint4*)(p);
  f.q[1] = *(const uint4*)(p + 16);
  return f.v;
}

// ---------------------------------------------------------------------------
// fp32 -> bf16 conversion (vectorized: float4 in, 4x bf16 out)
// ---------------------------------------------------------------------------
__global__ void k_cvt_bf16(const float* __restrict__ src, bf16_t* __restrict__ dst, int n4) {
  int i = blockIdx.x * blockDim.x + threadIdx.x;
  if (i >= n4) return;
  float4 f = ((const float4*)src)[i];
  union { bf16_t h[4]; uint2 u; } o;
  o.h[0] = (bf16_t)f.x; o.h[1] = (bf16_t)f.y;
  o.h[2] = (bf16_t)f.z; o.h[3] = (bf16_t)f.w;
  ((uint2*)dst)[i] = o.u;
}

// ---------------------------------------------------------------------------
// GEMM: C[m,n] = sum_k A[m,k] * W[n,k] + bias[n]   (A:[4096,1024], W:[1024,1024])
// Register-blocked: each wave owns a 32x64 output tile = 2 (M) x 4 (N)
// 16x16 accumulators -> 8 WMMAs per 32-deep k-step, 12 b128 loads.
// mode 0: -> Qh [B,H,T,DH] bf16     mode 1: -> Kh [B,H,T,DH] bf16
// mode 2: -> Vt [B,H,DH,T] bf16     mode 3: -> fp32 out [B*T, D]
// ---------------------------------------------------------------------------
__global__ void k_gemm_wmma(const bf16_t* __restrict__ A, const bf16_t* __restrict__ W,
                            const float* __restrict__ bias, bf16_t* __restrict__ outb,
                            float* __restrict__ outf, int mode) {
  const int lane = threadIdx.x & 31;
  const int w    = blockIdx.x * 8 + (threadIdx.x >> 5); // 2048 waves total
  const int ng   = w & 15;   // 0..15  (N/64)
  const int mt2  = w >> 4;   // 0..127 (M/32)

  const bf16_t* arow0 = A + (size_t)(mt2 * 32) * D_;
  const bf16_t* arow1 = arow0 + (size_t)16 * D_;
  const bf16_t* brow  = W + (size_t)(ng * 64) * D_;

  f32x8 acc[2][4] = {};
  for (int kk = 0; kk < D_; kk += 32) {
    bf16x16 a0 = load_fragA(arow0 + kk, D_, lane);
    bf16x16 a1 = load_fragA(arow1 + kk, D_, lane);
    bf16x16 b0 = load_fragB(brow + kk,                   D_, lane);
    bf16x16 b1 = load_fragB(brow + (size_t)16 * D_ + kk, D_, lane);
    bf16x16 b2 = load_fragB(brow + (size_t)32 * D_ + kk, D_, lane);
    bf16x16 b3 = load_fragB(brow + (size_t)48 * D_ + kk, D_, lane);
    acc[0][0] = WMMA_BF16(a0, b0, acc[0][0]);
    acc[0][1] = WMMA_BF16(a0, b1, acc[0][1]);
    acc[0][2] = WMMA_BF16(a0, b2, acc[0][2]);
    acc[0][3] = WMMA_BF16(a0, b3, acc[0][3]);
    acc[1][0] = WMMA_BF16(a1, b0, acc[1][0]);
    acc[1][1] = WMMA_BF16(a1, b1, acc[1][1]);
    acc[1][2] = WMMA_BF16(a1, b2, acc[1][2]);
    acc[1][3] = WMMA_BF16(a1, b3, acc[1][3]);
  }

  const int col   = lane & 15;
  const int rbase = (lane >> 4) << 3;      // rows 0..7 / 8..15 of each 16x16

  #pragma unroll
  for (int j = 0; j < 4; ++j) {
    const int n    = ng * 64 + j * 16 + col;
    const float bv = bias[n];
    const int h = n >> 6, d = n & 63;
    #pragma unroll
    for (int i = 0; i < 2; ++i) {
      #pragma unroll
      for (int r = 0; r < 8; ++r) {
        const int m  = mt2 * 32 + i * 16 + rbase + r;
        const int bb = m >> 11, t = m & (T_ - 1);
        const float val = acc[i][j][r] + bv;
        if (mode <= 1) {            // Qh / Kh : [B,H,T,DH]
          outb[(((size_t)(bb * H_ + h)) * T_ + t) * DH_ + d] = (bf16_t)val;
        } else if (mode == 2) {     // Vt : [B,H,DH,T]
          outb[(((size_t)(bb * H_ + h)) * DH_ + d) * T_ + t] = (bf16_t)val;
        } else {                    // final projection -> fp32 d_out
          outf[(size_t)m * D_ + n] = val;
        }
      }
    }
  }
}

// ---------------------------------------------------------------------------
// Scores + softmax: one wave per (b,h,q-tile of 16 rows).
// 16x2048 score strip lives in 128 KB LDS (CDNA5: 320 KB/WGP). Normalized
// probabilities streamed to d_out with non-temporal stores (537MB, no reuse
// window that fits in L2 -> keep Kh/Vt/weights resident instead).
// ---------------------------------------------------------------------------
#define SROW 2052   // 2048 + pad to break LDS bank aliasing between row halves

__global__ void k_scores_softmax(const bf16_t* __restrict__ Qh, const bf16_t* __restrict__ Kh,
                                 float* __restrict__ attn) {
  extern __shared__ float s[];             // [16][SROW]
  const int lane = threadIdx.x;            // 32 threads = 1 wave
  const int tile = blockIdx.x;             // bh*128 + qt
  const int qt   = tile & 127;
  const int bh   = tile >> 7;              // b*H + h

  const bf16_t* qb = Qh + ((size_t)bh * T_ + qt * 16) * DH_;
  bf16x16 a0 = load_fragA(qb,      DH_, lane);   // k = 0..31
  bf16x16 a1 = load_fragA(qb + 32, DH_, lane);   // k = 32..63

  const bf16_t* kb0 = Kh + (size_t)bh * T_ * DH_;
  const int col   = lane & 15;
  const int rbase = (lane >> 4) << 3;

  for (int kt = 0; kt < T_ / 16; ++kt) {
    const bf16_t* kb = kb0 + (size_t)kt * 16 * DH_;
    bf16x16 b0 = load_fragB(kb,      DH_, lane);
    bf16x16 b1 = load_fragB(kb + 32, DH_, lane);
    f32x8 acc = {};
    acc = WMMA_BF16(a0, b0, acc);
    acc = WMMA_BF16(a1, b1, acc);
    #pragma unroll
    for (int r = 0; r < 8; ++r)
      s[(rbase + r) * SROW + kt * 16 + col] = acc[r] * 0.125f;   // 1/sqrt(64)
  }
  __syncthreads();

  float* orow = attn + ((size_t)bh * T_ + (size_t)qt * 16) * T_;
  for (int m = 0; m < 16; ++m) {
    float mx = -3.402823e38f;
    for (int j = lane; j < T_; j += 32) mx = fmaxf(mx, s[m * SROW + j]);
    #pragma unroll
    for (int o = 16; o > 0; o >>= 1) mx = fmaxf(mx, __shfl_xor(mx, o, 32));
    float sum = 0.f;
    for (int j = lane; j < T_; j += 32) {
      float e = __expf(s[m * SROW + j] - mx);
      s[m * SROW + j] = e;
      sum += e;
    }
    #pragma unroll
    for (int o = 16; o > 0; o >>= 1) sum += __shfl_xor(sum, o, 32);
    const float inv = 1.0f / sum;
    for (int j = lane; j < T_; j += 32)
      __builtin_nontemporal_store(s[m * SROW + j] * inv, &orow[(size_t)m * T_ + j]);
  }
}

// ---------------------------------------------------------------------------
// attn @ V : per wave one (b,h, 16-row q-tile) x ALL 4 d-tiles (16x64 out).
// A = fp32 attn streamed with non-temporal loads, cvt->bf16 in-register,
// amortized over 4 WMMAs. B^T = Vt (K-contiguous bf16, L2-resident).
// Output merged-head bf16 xh [B*T, D].
// ---------------------------------------------------------------------------
__global__ void k_attn_v(const float* __restrict__ attn, const bf16_t* __restrict__ Vt,
                         bf16_t* __restrict__ xh) {
  const int lane = threadIdx.x & 31;
  const int w    = blockIdx.x * 8 + (threadIdx.x >> 5); // 4096 waves
  const int qt   = w & 127;          // T/16
  const int bh   = w >> 7;           // b*H + h

  const float*  abase = attn + ((size_t)bh * T_ + qt * 16) * T_;
  const bf16_t* vbase = Vt   + (size_t)bh * DH_ * T_;     // rows d (64), cols t

  const int r   = lane & 15;
  const int koA = (lane < 16) ? 0 : 8;
  const float* arow = abase + (size_t)r * T_ + koA;

  f32x8 acc[4] = {};
  for (int kt = 0; kt < T_; kt += 32) {
    f32x4 f0 = __builtin_nontemporal_load((const f32x4*)(arow + kt));
    f32x4 f1 = __builtin_nontemporal_load((const f32x4*)(arow + kt + 4));
    f32x4 f2 = __builtin_nontemporal_load((const f32x4*)(arow + kt + 16));
    f32x4 f3 = __builtin_nontemporal_load((const f32x4*)(arow + kt + 20));
    FragU fa;
    #pragma unroll
    for (int e = 0; e < 4; ++e) {
      fa.v[e]      = (bf16_t)f0[e];
      fa.v[4 + e]  = (bf16_t)f1[e];
      fa.v[8 + e]  = (bf16_t)f2[e];
      fa.v[12 + e] = (bf16_t)f3[e];
    }
    bf16x16 b0 = load_fragB(vbase + kt,                   T_, lane);
    bf16x16 b1 = load_fragB(vbase + (size_t)16 * T_ + kt, T_, lane);
    bf16x16 b2 = load_fragB(vbase + (size_t)32 * T_ + kt, T_, lane);
    bf16x16 b3 = load_fragB(vbase + (size_t)48 * T_ + kt, T_, lane);
    acc[0] = WMMA_BF16(fa.v, b0, acc[0]);
    acc[1] = WMMA_BF16(fa.v, b1, acc[1]);
    acc[2] = WMMA_BF16(fa.v, b2, acc[2]);
    acc[3] = WMMA_BF16(fa.v, b3, acc[3]);
  }

  const int col   = lane & 15;
  const int rbase = (lane >> 4) << 3;
  const int bb    = bh >> 4;
  const int h     = bh & 15;
  #pragma unroll
  for (int j = 0; j < 4; ++j) {
    const int n = h * DH_ + j * 16 + col;
    #pragma unroll
    for (int rr = 0; rr < 8; ++rr) {
      const int t = qt * 16 + rbase + rr;
      xh[((size_t)bb * T_ + t) * D_ + n] = (bf16_t)acc[j][rr];
    }
  }
}

// ---------------------------------------------------------------------------
extern "C" void kernel_launch(void* const* d_in, const int* in_sizes, int n_in,
                              void* d_out, int out_size, void* d_ws, size_t ws_size,
                              hipStream_t stream) {
  const float* q  = (const float*)d_in[0];
  const float* k  = (const float*)d_in[1];
  const float* v  = (const float*)d_in[2];
  const float* Wq = (const float*)d_in[3];
  const float* bq = (const float*)d_in[4];
  const float* Wk = (const float*)d_in[5];
  const float* bk = (const float*)d_in[6];
  const float* Wv = (const float*)d_in[7];
  const float* bv = (const float*)d_in[8];
  const float* Wo = (const float*)d_in[9];
  const float* bo = (const float*)d_in[10];

  float* outp = (float*)d_out;                       // [B*T, D]
  float* attn = (float*)d_out + (size_t)M_ * D_;     // [B,H,T,T]

  // Workspace layout (bf16 elements): 64 MB total
  bf16_t* ws = (bf16_t*)d_ws;
  const size_t NX = (size_t)M_ * D_;   // 4,194,304
  const size_t NW = (size_t)D_ * D_;   // 1,048,576
  bf16_t* xq = ws;
  bf16_t* xk = xq + NX;
  bf16_t* xv = xk + NX;
  bf16_t* wq = xv + NX;
  bf16_t* wk = wq + NW;
  bf16_t* wv = wk + NW;
  bf16_t* wo = wv + NW;
  bf16_t* Qh = wo + NW;
  bf16_t* Kh = Qh + NX;
  bf16_t* Vt = Kh + NX;
  bf16_t* xh = Vt + NX;

  // 1) fp32 -> bf16 conversions
  k_cvt_bf16<<<(int)(NX / 4 / 256), 256, 0, stream>>>(q,  xq, (int)(NX / 4));
  k_cvt_bf16<<<(int)(NX / 4 / 256), 256, 0, stream>>>(k,  xk, (int)(NX / 4));
  k_cvt_bf16<<<(int)(NX / 4 / 256), 256, 0, stream>>>(v,  xv, (int)(NX / 4));
  k_cvt_bf16<<<(int)(NW / 4 / 256), 256, 0, stream>>>(Wq, wq, (int)(NW / 4));
  k_cvt_bf16<<<(int)(NW / 4 / 256), 256, 0, stream>>>(Wk, wk, (int)(NW / 4));
  k_cvt_bf16<<<(int)(NW / 4 / 256), 256, 0, stream>>>(Wv, wv, (int)(NW / 4));
  k_cvt_bf16<<<(int)(NW / 4 / 256), 256, 0, stream>>>(Wo, wo, (int)(NW / 4));

  // 2) Q/K/V projections (register-blocked WMMA bf16), V stored transposed
  const int gemmBlocks = (M_ / 32) * (D_ / 64) / 8;   // 256 blocks x 8 waves
  k_gemm_wmma<<<gemmBlocks, 256, 0, stream>>>(xq, wq, bq, Qh, nullptr, 0);
  k_gemm_wmma<<<gemmBlocks, 256, 0, stream>>>(xk, wk, bk, Kh, nullptr, 1);
  k_gemm_wmma<<<gemmBlocks, 256, 0, stream>>>(xv, wv, bv, Vt, nullptr, 2);

  // 3) scores + softmax (128 KB dynamic LDS per workgroup)
  const int ldsBytes = 16 * SROW * (int)sizeof(float);   // 131,328
  (void)hipFuncSetAttribute((const void*)k_scores_softmax,
                            hipFuncAttributeMaxDynamicSharedMemorySize, ldsBytes);
  k_scores_softmax<<<B_ * H_ * (T_ / 16), 32, ldsBytes, stream>>>(Qh, Kh, attn);

  // 4) attn @ V -> merged-head bf16 (16x64 per wave)
  k_attn_v<<<B_ * H_ * (T_ / 16) / 8, 256, 0, stream>>>(attn, Vt, xh);

  // 5) output projection -> fp32 d_out
  k_gemm_wmma<<<gemmBlocks, 256, 0, stream>>>(xh, wo, bo, nullptr, outp, 3);
}
